// ScycloneWaveRNN_22832046145710
// MI455X (gfx1250) — matrix operand order repaired and checked by reading, not compiled, not run on hardware
//
#include <hip/hip_runtime.h>
#include <hip/hip_bf16.h>

// ============================================================================
// ScycloneWaveRNN on MI455X (gfx1250, wave32, WMMA)
//
// Phase 1 (compute-bound, ~181 GFLOP): MLP 64->512->1024->2048->4096 on 8192
//   rows via V_WMMA_F32_16X16X32_F16. Wave tile 16Mx64N (4 accumulators, A
//   fragment reused 4x), block tile 32Mx256N (8 waves).
// Phase 2 (latency-bound): 4096-step GRU scan on 16 WGPs. Fused [Whh|Wih]
//   weights (2.25 MB f16) stay resident in the 192 MB L2. feat_{t+1} (16 KB)
//   is prefetched into double-buffered LDS with GLOBAL_LOAD_ASYNC_TO_LDS_B128
//   during step t, retired with s_wait_asynccnt 4 (async loads complete in
//   issue order, so threshold 4 waits only for the older buffer). Cross-WGP
//   step sync = CDNA5 cluster barrier + resident-grid atomic counter. 2-wide
//   output head reduced with global_atomic_add_f32 into triple-buffered slots.
// ============================================================================

typedef __attribute__((ext_vector_type(16))) _Float16 v16h;
typedef __attribute__((ext_vector_type(8)))  _Float16 v8h;
typedef __attribute__((ext_vector_type(8)))  float    v8f;

__device__ __forceinline__ v8f wmma_f16(v16h a, v16h b, v8f c) {
  return __builtin_amdgcn_wmma_f32_16x16x32_f16(
      /*neg_a=*/false, a, /*neg_b=*/false, b,
      /*c_mod=*/(short)0, c, /*reuse_a=*/false, /*reuse_b=*/false);
}

// A fragment: 16x32 f16, A row-major with row stride `lda` (elements).
// Lanes 0-15: M=lane, K=0..15; lanes 16-31: M=lane-16,
// K=16..31  (contiguous order, matches load_bfrag).
__device__ __forceinline__ v16h load_afrag(const _Float16* A, long lda,
                                           int m0, int k0) {
  int lane = threadIdx.x & 31;
  int half = lane >> 4;
  int mr   = lane & 15;
  const _Float16* p = A + (long)(m0 + mr) * lda + k0 + half * 16;
  v8h lo = *(const v8h*)p;         // K = k0 + half*16 .. +7
  v8h hi = *(const v8h*)(p + 8);   // K = k0 + half*16 + 8 .. +7
  v16h a;
  #pragma unroll
  for (int i = 0; i < 8; ++i) { a[i] = lo[i]; a[i + 8] = hi[i]; }
  return a;
}

// B fragment: 32x16 (K x N) from row-major W (N x K, row stride ldb).
// Lanes 0-15: N=lane, K=k0..k0+15; lanes 16-31: N=lane-16, K=k0+16..31
// -> single 32-byte v16h load per lane.
__device__ __forceinline__ v16h load_bfrag(const _Float16* W, long ldb,
                                           int n0, int k0) {
  int lane = threadIdx.x & 31;
  int half = lane >> 4;
  int nr   = lane & 15;
  return *(const v16h*)(W + (long)(n0 + nr) * ldb + k0 + half * 16);
}

// ---------------------------------------------------------------------------
// Elementwise helpers
// ---------------------------------------------------------------------------
__global__ void cast_f16_kernel(const float* __restrict__ in,
                                _Float16* __restrict__ out, int n) {
  int i = blockIdx.x * blockDim.x + threadIdx.x;
  if (i < n) out[i] = (_Float16)in[i];
}

// Fused RNN weight: Wf[g][0:512] = Whh[g][:], Wf[g][512:768] = Wih[g][0:256]
// (Wih row length is 257; col 256 is w_last, applied in the scan epilogue).
__global__ void build_fused_kernel(const float* __restrict__ Whh,
                                   const float* __restrict__ Wih,
                                   _Float16* __restrict__ Wf) {
  int i = blockIdx.x * blockDim.x + threadIdx.x;
  if (i >= 1536 * 768) return;
  int g = i / 768, k = i % 768;
  float v = (k < 512) ? Whh[g * 512 + k] : Wih[g * 257 + (k - 512)];
  Wf[i] = (_Float16)v;
}

__global__ void init_state_kernel(_Float16* hbufs, float* obuf,
                                  const float* __restrict__ bb,
                                  unsigned* ctr) {
  int i = blockIdx.x * blockDim.x + threadIdx.x;
  if (i < 2 * 32 * 512) hbufs[i] = (_Float16)0.0f;     // hprev double buffer
  if (i < 3 * 64)       obuf[i]  = bb[i & 1];          // o slots seeded w/ bias
  if (i == 0)           *ctr     = 0u;
}

// ---------------------------------------------------------------------------
// WMMA GEMM: C(MxN,f16) = relu(A(MxK,f16) @ W(NxK,f16)^T + bias)
// 256 threads = 8 waves arranged 2(M) x 4(N); wave tile 16x64 (4 accums,
// A fragment reused 4x per k-step); block tile 32x256.
// ---------------------------------------------------------------------------
__global__ void __launch_bounds__(256)
gemm_bias_relu_kernel(const _Float16* __restrict__ A,
                      const _Float16* __restrict__ W,
                      const float* __restrict__ bias,
                      _Float16* __restrict__ C, int M, int N, int K) {
  int wave = threadIdx.x >> 5;
  int mt = wave >> 2;                       // 0..1
  int ng = wave & 3;                        // 0..3
  int m0 = blockIdx.y * 32 + mt * 16;
  int n0 = blockIdx.x * 256 + ng * 64;
  v8f acc[4] = {{}, {}, {}, {}};
  for (int k0 = 0; k0 < K; k0 += 32) {
    v16h a = load_afrag(A, K, m0, k0);
    #pragma unroll
    for (int j = 0; j < 4; ++j) {
      v16h b = load_bfrag(W, K, n0 + j * 16, k0);
      acc[j] = wmma_f16(a, b, acc[j]);
    }
  }
  int lane = threadIdx.x & 31, half = lane >> 4, n = lane & 15;
  int mb = m0 + half * 8;
  #pragma unroll
  for (int j = 0; j < 4; ++j) {
    float bn = bias[n0 + j * 16 + n];
    #pragma unroll
    for (int r = 0; r < 8; ++r) {
      float v = acc[j][r] + bn;
      v = v > 0.0f ? v : 0.0f;
      C[(long)(mb + r) * N + n0 + j * 16 + n] = (_Float16)v;
    }
  }
}

// ---------------------------------------------------------------------------
// Grid-wide sync for the 16-block scan: CDNA5 cluster barrier (NOP when not
// dispatched as a cluster) + release/acquire atomic counter (16 blocks are
// all resident, so the spin is deadlock-free).
// ---------------------------------------------------------------------------
__device__ __forceinline__ void grid_sync(unsigned* ctr, unsigned target) {
  __threadfence();
  __syncthreads();
  if (threadIdx.x < 32) {
    __builtin_amdgcn_s_cluster_barrier();   // s_barrier_signal/-wait -3
  }
  if (threadIdx.x == 0) {
    __hip_atomic_fetch_add(ctr, 1u, __ATOMIC_ACQ_REL, __HIP_MEMORY_SCOPE_AGENT);
    while (__hip_atomic_load(ctr, __ATOMIC_ACQUIRE, __HIP_MEMORY_SCOPE_AGENT)
           < target) {
      __builtin_amdgcn_s_sleep(1);
    }
  }
  __syncthreads();
  __threadfence();
}

// ---------------------------------------------------------------------------
// Persistent GRU scan. 16 blocks x 256 threads (8 waves). Block wg owns
// hidden units [wg*32, wg*32+32): 12 gate-tile WMMA jobs/step (K=768, split
// accumulators for the n-gate), 4 head-tile jobs (K=512). feat tiles arrive
// via async-to-LDS prefetch one step ahead.
// ---------------------------------------------------------------------------
#define FEAT_LD 264   // 256 + 8 halves pad: lanes 0-15 hit disjoint bank quads

__global__ void __launch_bounds__(256)
scan_kernel(const _Float16* __restrict__ h4,    // (32,4096,256) f16 features
            const _Float16* __restrict__ Wf,    // fused (1536,768) f16
            const _Float16* __restrict__ Wah,   // (512,512) f16
            const float* __restrict__ bih, const float* __restrict__ bhh,
            const float* __restrict__ Wih,      // f32, for w_last = col 256
            const float* __restrict__ ba,  const float* __restrict__ Wb,
            const float* __restrict__ bb,  const float* __restrict__ noise,
            _Float16* hbufs,                    // 2 x (32x512) f16
            float* obuf,                        // 3 x (32x2) f32
            unsigned* ctr, float* out) {
  __shared__ float lds_r[32][32], lds_z[32][32];
  __shared__ float lds_hn[32][32], lds_in[32][32];
  __shared__ float lds_h[32][32];               // owned hprev, persists over t
  __shared__ float lds_a[32][32];               // head activations
  __shared__ float lds_prev[32];                // previous sample per batch
  __shared__ _Float16 lds_feat[2][32][FEAT_LD]; // double-buffered feat tile

  const int wg = blockIdx.x;                    // 0..15, one per WGP
  const int j0 = wg * 32;
  const int tid = threadIdx.x;
  const int wave = tid >> 5;

  for (int i = tid; i < 1024; i += 256) lds_h[i >> 5][i & 31] = 0.0f;
  if (tid < 32) lds_prev[tid] = 0.0f;
  __syncthreads();

  // Async prefetch of one feat tile (32 rows x 512 B) into LDS: 1024 16-byte
  // segments over 256 lanes -> 4 GLOBAL_LOAD_ASYNC_TO_LDS_B128 per wave.
  auto prefetch_feat = [&](int t, int buf) {
    #pragma unroll
    for (int s = 0; s < 4; ++s) {
      int idx = tid + s * 256;                  // 0..1023
      int row = idx >> 5, seg = idx & 31;
      const _Float16* g =
          h4 + (long)row * 1048576 + (long)t * 256 + seg * 8;
      unsigned ldso =
          (unsigned)(unsigned long long)(&lds_feat[buf][row][seg * 8]);
      asm volatile("global_load_async_to_lds_b128 %0, %1, off"
                   :: "v"(ldso), "v"(g) : "memory");
    }
  };

  prefetch_feat(0, 0);                          // warm buffer for t=0

  unsigned barno = 0;
  for (int t = 0; t < 4096; ++t) {
    const _Float16* hp     = hbufs + (size_t)(t & 1) * (32 * 512);
    _Float16*       hn_out = hbufs + (size_t)((t + 1) & 1) * (32 * 512);

    // Kick off t+1 prefetch, then wait only for the OLDER 4 async loads
    // (issue-order completion) so buffer t&1 is ready while t+1 streams.
    int tpre = (t + 1 < 4096) ? t + 1 : 4095;
    prefetch_feat(tpre, (t + 1) & 1);
    asm volatile("s_wait_asynccnt 0x4" ::: "memory");
    __syncthreads();
    const _Float16* featA = &lds_feat[t & 1][0][0];

    // ---- gate GEMMs: 12 jobs of 16x16, K = 512 (hidden) + 256 (input) ----
    for (int jj = wave; jj < 12; jj += 8) {
      int mt = jj / 6, rem = jj % 6, gate = rem >> 1, nt = rem & 1;
      int m0 = mt * 16;
      int g0 = gate * 512 + j0 + nt * 16;
      v8f acch = {}, acci = {};
      for (int k0 = 0; k0 < 512; k0 += 32) {
        v16h a = load_afrag(hp, 512, m0, k0);
        v16h b = load_bfrag(Wf, 768, g0, k0);
        acch = wmma_f16(a, b, acch);
      }
      for (int k0 = 0; k0 < 256; k0 += 32) {    // input part from LDS
        v16h a = load_afrag(featA, FEAT_LD, m0, k0);
        v16h b = load_bfrag(Wf, 768, g0, 512 + k0);
        acci = wmma_f16(a, b, acci);
      }
      int lane = tid & 31, half = lane >> 4, n = lane & 15;
      int jl = nt * 16 + n;
      int mb = m0 + half * 8;
      int g = g0 + n;
      float bi = bih[g], bh = bhh[g], wl = Wih[g * 257 + 256];
      #pragma unroll
      for (int r = 0; r < 8; ++r) {
        int m = mb + r;
        float pv = lds_prev[m];
        if (gate == 2) {                        // n-gate: keep halves separate
          lds_hn[m][jl] = acch[r] + bh;
          lds_in[m][jl] = acci[r] + bi + pv * wl;
        } else {
          float v = acch[r] + acci[r] + bi + bh + pv * wl;
          if (gate == 0) lds_r[m][jl] = v; else lds_z[m][jl] = v;
        }
      }
    }
    __syncthreads();

    // ---- GRU cell update for owned 32 hidden units, all 32 batches ----
    for (int i = tid; i < 1024; i += 256) {
      int m = i >> 5, jl = i & 31;
      float r  = 1.0f / (1.0f + __expf(-lds_r[m][jl]));
      float z  = 1.0f / (1.0f + __expf(-lds_z[m][jl]));
      float nn = tanhf(lds_in[m][jl] + r * lds_hn[m][jl]);
      float hprev = lds_h[m][jl];
      float hnew  = (1.0f - z) * nn + z * hprev;
      lds_h[m][jl] = hnew;
      hn_out[m * 512 + j0 + jl] = (_Float16)hnew;   // broadcast to cluster
    }

    grid_sync(ctr, 16u * (++barno));                // hnew visible everywhere

    // reset o slot for step t+2 (last read at t-1; ordered by this barrier)
    if (wg == 0 && tid < 64) obuf[((t + 2) % 3) * 64 + tid] = bb[tid & 1];

    // ---- head: a = relu(hnew @ Wa^T + ba), owned 32 columns, K=512 ----
    if (wave < 4) {
      int mt = wave >> 1, nt = wave & 1;
      int m0 = mt * 16, c0 = j0 + nt * 16;
      v8f acc = {};
      for (int k0 = 0; k0 < 512; k0 += 32) {
        v16h a = load_afrag(hn_out, 512, m0, k0);
        v16h b = load_bfrag(Wah, 512, c0, k0);
        acc = wmma_f16(a, b, acc);
      }
      int lane = tid & 31, half = lane >> 4, n = lane & 15;
      int mb = m0 + half * 8;
      float bav = ba[c0 + n];
      #pragma unroll
      for (int r = 0; r < 8; ++r) {
        float v = acc[r] + bav;
        lds_a[mb + r][nt * 16 + n] = v > 0.0f ? v : 0.0f;
      }
    }
    __syncthreads();

    // ---- partial o = a_slice @ Wb_slice^T, reduced across cluster ----
    if (tid < 64) {
      int m = tid >> 1, oi = tid & 1;
      const float* wbr = Wb + oi * 512 + j0;
      float s = 0.0f;
      #pragma unroll
      for (int jl = 0; jl < 32; ++jl) s += lds_a[m][jl] * wbr[jl];
      atomicAdd(&obuf[(t % 3) * 64 + m * 2 + oi], s);  // global_atomic_add_f32
    }

    grid_sync(ctr, 16u * (++barno));                // o fully reduced

    // ---- sample; every block computes prev locally, wg 0 writes outputs ----
    if (tid < 32) {
      int m = tid;
      float mv = obuf[(t % 3) * 64 + m * 2 + 0];
      float sv = obuf[(t % 3) * 64 + m * 2 + 1];
      float sp = mv + __expf(sv) * noise[t * 32 + m];
      lds_prev[m] = sp;
      if (wg == 0) {
        out[(long)m * 4096 + t]          = mv;   // m.T
        out[131072 + (long)m * 4096 + t] = sv;   // s.T
        out[262144 + (long)m * 4096 + t] = sp;   // samp.T
      }
    }
    __syncthreads();
  }
}

// ---------------------------------------------------------------------------
extern "C" void kernel_launch(void* const* d_in, const int* in_sizes, int n_in,
                              void* d_out, int out_size, void* d_ws,
                              size_t ws_size, hipStream_t stream) {
  const float* x    = (const float*)d_in[0];
  const float* nois = (const float*)d_in[1];
  const float* W1   = (const float*)d_in[2];
  const float* b1   = (const float*)d_in[3];
  const float* W2   = (const float*)d_in[4];
  const float* b2   = (const float*)d_in[5];
  const float* W3   = (const float*)d_in[6];
  const float* b3   = (const float*)d_in[7];
  const float* W4   = (const float*)d_in[8];
  const float* b4   = (const float*)d_in[9];
  const float* Wih  = (const float*)d_in[10];
  const float* bih  = (const float*)d_in[11];
  const float* Whh  = (const float*)d_in[12];
  const float* bhh  = (const float*)d_in[13];
  const float* Wa   = (const float*)d_in[14];
  const float* ba   = (const float*)d_in[15];
  const float* Wb   = (const float*)d_in[16];
  const float* bb   = (const float*)d_in[17];
  float* out = (float*)d_out;

  char* p = (char*)d_ws;
  auto alloc = [&](size_t bytes) -> char* {
    char* r = p;
    p += (bytes + 255) & ~(size_t)255;
    return r;
  };
  _Float16* xh    = (_Float16*)alloc((size_t)524288 * 2);
  _Float16* W1h   = (_Float16*)alloc((size_t)32768 * 2);
  _Float16* W2h   = (_Float16*)alloc((size_t)524288 * 2);
  _Float16* W3h   = (_Float16*)alloc((size_t)2097152 * 2);
  _Float16* W4h   = (_Float16*)alloc((size_t)8388608 * 2);
  _Float16* Wfh   = (_Float16*)alloc((size_t)1536 * 768 * 2);
  _Float16* Wah   = (_Float16*)alloc((size_t)262144 * 2);
  _Float16* h1    = (_Float16*)alloc((size_t)8192 * 512 * 2);
  _Float16* h2    = (_Float16*)alloc((size_t)8192 * 1024 * 2);
  _Float16* h3    = (_Float16*)alloc((size_t)8192 * 2048 * 2);
  _Float16* h4    = (_Float16*)alloc((size_t)8192 * 4096 * 2);
  _Float16* hbufs = (_Float16*)alloc((size_t)2 * 32 * 512 * 2);
  float*    obuf  = (float*)alloc((size_t)3 * 64 * 4);
  unsigned* ctr   = (unsigned*)alloc(256);

  auto cast = [&](const float* src, _Float16* dst, int n) {
    cast_f16_kernel<<<(n + 255) / 256, 256, 0, stream>>>(src, dst, n);
  };
  cast(x,  xh,  524288);
  cast(W1, W1h, 32768);
  cast(W2, W2h, 524288);
  cast(W3, W3h, 2097152);
  cast(W4, W4h, 8388608);
  cast(Wa, Wah, 262144);
  build_fused_kernel<<<(1536 * 768 + 255) / 256, 256, 0, stream>>>(Whh, Wih,
                                                                   Wfh);

  // MLP: M=8192 rows, block tile 32x256, wave tile 16x64
  gemm_bias_relu_kernel<<<dim3(512 / 256, 8192 / 32), 256, 0, stream>>>(
      xh, W1h, b1, h1, 8192, 512, 64);
  gemm_bias_relu_kernel<<<dim3(1024 / 256, 8192 / 32), 256, 0, stream>>>(
      h1, W2h, b2, h2, 8192, 1024, 512);
  gemm_bias_relu_kernel<<<dim3(2048 / 256, 8192 / 32), 256, 0, stream>>>(
      h2, W3h, b3, h3, 8192, 2048, 1024);
  gemm_bias_relu_kernel<<<dim3(4096 / 256, 8192 / 32), 256, 0, stream>>>(
      h3, W4h, b4, h4, 8192, 4096, 2048);

  init_state_kernel<<<128, 256, 0, stream>>>(hbufs, obuf, bb, ctr);

  scan_kernel<<<16, 256, 0, stream>>>(h4, Wfh, Wah, bih, bhh, Wih, ba, Wb, bb,
                                      nois, hbufs, obuf, ctr, out);
}
